// ClassicalMHA_3710851744123
// MI455X (gfx1250) — compile-verified, matrix operand order checked by reading, not run on hardware
//
#include <hip/hip_runtime.h>
#include <hip/hip_bf16.h>

// ===================== CDNA5 (gfx1250) fused causal MHA =====================
// fp32 in/out; internally f16 via v_wmma_f32_16x16x32_f16 (f32 accumulate).
// Flash-attention style: no [T,T] score materialization.

typedef __attribute__((ext_vector_type(16))) _Float16 v16h;
typedef __attribute__((ext_vector_type(8)))  float    v8f;

#define D_MODEL  1024
#define SEQ_T    2048
#define BATCH    2
#define HEADS    16
#define HEAD_DIM 64
#define BT       (BATCH * SEQ_T)          /* 4096 rows */
#define ATT_SCALE 0.125f                  /* 1/sqrt(64) */
#define NEG_BIG  (-1.0e30f)

// ---------------------------------------------------------------------------
// Fragment loader for V_WMMA_F32_16X16X32_F16.
// 16-bit A (16x32) layout (ISA 7.12.2): lane = half*16 + row; per lane the 16
// halves are two contiguous 8-half chunks at K = kbase + half*8 and
// K = kbase + 16 + half*8.  The B (32x16) fragment uses the same per-lane K
// striping with "row" = its N column, so one loader serves A, and B when the
// source is stored K-contiguous (i.e. transposed weights / row-major K / V^T).
// ---------------------------------------------------------------------------
__device__ __forceinline__ v16h load_frag16(const _Float16* __restrict__ p,
                                            int ld, int row, int kbase, int half) {
    union { v16h v; uint4 q[2]; } u;
    const _Float16* r0 = p + (size_t)row * ld + kbase + half * 8;
    u.q[0] = *(const uint4*)(r0);
    u.q[1] = *(const uint4*)(r0 + 16);
    return u.v;
}

__device__ __forceinline__ v8f wmma16(v16h a, v16h b, v8f c) {
    // D = A(16x32 f16) * B(32x16 f16) + C(16x16 f32)
    return __builtin_amdgcn_wmma_f32_16x16x32_f16(
        /*neg_a=*/false, a, /*neg_b=*/false, b,
        /*c_mod=*/(short)0, c, /*reuse_a=*/false, /*reuse_b=*/false);
}

// ---------------------------------------------------------------------------
// Prep: f32 -> f16 copy of x
// ---------------------------------------------------------------------------
__global__ void mha_convert_x(const float* __restrict__ x, _Float16* __restrict__ xh) {
    int i = blockIdx.x * 256 + threadIdx.x;
    xh[i] = (_Float16)x[i];
}

// Prep: W[in,out] -> W^T[out,in] in f16 (so B-fragments are K-contiguous)
__global__ void mha_transpose_w(const float* __restrict__ Wq, const float* __restrict__ Wk,
                                const float* __restrict__ Wv, const float* __restrict__ Wo,
                                _Float16* __restrict__ Tq, _Float16* __restrict__ Tk,
                                _Float16* __restrict__ Tv, _Float16* __restrict__ To) {
    int idx = blockIdx.x * 256 + threadIdx.x;      // idx = o*1024 + i
    int o = idx >> 10;
    int i = idx & 1023;
    const float* W; _Float16* T;
    switch (blockIdx.y) {
        case 0:  W = Wq; T = Tq; break;
        case 1:  W = Wk; T = Tk; break;
        case 2:  W = Wv; T = Tv; break;
        default: W = Wo; T = To; break;
    }
    T[idx] = (_Float16)W[(size_t)i * D_MODEL + o];
}

// ---------------------------------------------------------------------------
// GEMM: C[BT x D] = A_h[BT x D] * W (via W^T f16) + bias
// Wave tile 16x64, 8 waves/block (256 thr) stacked in M -> block tile 128x64.
// mode 0: store f16 head layout   [B,H,T,Hd]          (Q, K)
// mode 1: store f16 head-T layout [B,H,Hd,T]          (V, for contiguous P*V)
// mode 2: store f32 row-major [BT, D]                  (final output + bias)
// ---------------------------------------------------------------------------
__global__ void mha_gemm(const _Float16* __restrict__ A,
                         const _Float16* __restrict__ WT,
                         const float* __restrict__ bias,
                         int mode,
                         _Float16* __restrict__ outh,
                         float* __restrict__ outf) {
    const int lane = threadIdx.x & 31;
    const int w    = __builtin_amdgcn_readfirstlane((int)(threadIdx.x >> 5));
    const int half = lane >> 4;
    const int nc   = lane & 15;

    const int m0 = blockIdx.x * 128 + w * 16;
    const int n0 = blockIdx.y * 64;

    v8f acc[4];
    #pragma unroll
    for (int t = 0; t < 4; ++t) acc[t] = v8f{};

    for (int k = 0; k < D_MODEL; k += 32) {
        if (k + 32 < D_MODEL)
            __builtin_prefetch(A + (size_t)(m0 + nc) * D_MODEL + k + 32, 0, 0);
        v16h a = load_frag16(A, D_MODEL, m0 + nc, k, half);
        #pragma unroll
        for (int t = 0; t < 4; ++t) {
            v16h b = load_frag16(WT, D_MODEL, n0 + t * 16 + nc, k, half);
            acc[t] = wmma16(a, b, acc[t]);
        }
    }

    // Epilogue: C/D layout -> element (m = r + 8*half, n = nc) per VGPR r.
    #pragma unroll
    for (int t = 0; t < 4; ++t) {
        const int col = n0 + t * 16 + nc;
        const float bv = bias[col];
        const int hh = col >> 6;       // head
        const int hd = col & 63;       // index within head
        #pragma unroll
        for (int r = 0; r < 8; ++r) {
            const int row = m0 + r + 8 * half;       // token row in [0, BT)
            const float v = acc[t][r] + bv;
            const int bb = row >> 11;                // batch
            const int tt = row & 2047;               // position
            if (mode == 0) {
                outh[(((size_t)(bb * HEADS + hh)) * SEQ_T + tt) * HEAD_DIM + hd] = (_Float16)v;
            } else if (mode == 1) {
                outh[(((size_t)(bb * HEADS + hh)) * HEAD_DIM + hd) * SEQ_T + tt] = (_Float16)v;
            } else {
                outf[(size_t)row * D_MODEL + col] = v;
            }
        }
    }
}

// ---------------------------------------------------------------------------
// Flash attention (causal, online softmax).
// One wave owns a 16-row Q tile; 8 waves/block -> 128 query rows per block.
// Q: [B,H,T,Hd] f16; K: [B,H,T,Hd] f16; V: [B,H,Hd,T] f16 (transposed).
// Output AO: f16 [BT, D] row-major (head-interleaved) for the final GEMM.
// ---------------------------------------------------------------------------
__global__ void mha_attention(const _Float16* __restrict__ Qh,
                              const _Float16* __restrict__ Kh,
                              const _Float16* __restrict__ VT,
                              _Float16* __restrict__ AO) {
    const int lane = threadIdx.x & 31;
    const int w    = __builtin_amdgcn_readfirstlane((int)(threadIdx.x >> 5));
    const int half = lane >> 4;
    const int nc   = lane & 15;

    const int h  = blockIdx.y;
    const int b  = blockIdx.z;
    const int q0 = blockIdx.x * 128 + w * 16;

    const _Float16* Qp = Qh + (size_t)(b * HEADS + h) * SEQ_T * HEAD_DIM;
    const _Float16* Kp = Kh + (size_t)(b * HEADS + h) * SEQ_T * HEAD_DIM;
    const _Float16* Vp = VT + (size_t)(b * HEADS + h) * HEAD_DIM * SEQ_T;

    // Per-wave LDS staging for P (16x64 f16) to convert D-layout -> A-layout.
    __shared__ __align__(16) _Float16 Pst[8][16 * 64];
    _Float16* pw = &Pst[w][0];

    // Q tile is reused across every KV step: preload both K-chunks.
    const v16h qa0 = load_frag16(Qp, HEAD_DIM, q0 + nc, 0,  half);
    const v16h qa1 = load_frag16(Qp, HEAD_DIM, q0 + nc, 32, half);

    v8f acc[4];
    #pragma unroll
    for (int t = 0; t < 4; ++t) acc[t] = v8f{};
    float mrun[8], lrun[8];
    #pragma unroll
    for (int r = 0; r < 8; ++r) { mrun[r] = NEG_BIG; lrun[r] = 0.0f; }

    const int kv_end = q0 + 16;                      // causal: keys < q0+16
    for (int kv0 = 0; kv0 < kv_end; kv0 += 64) {
        // ---- S = Q K^T (16 x 64) ----
        v8f s[4];
        #pragma unroll
        for (int j = 0; j < 4; ++j) {
            s[j] = v8f{};
            v16h kb0 = load_frag16(Kp, HEAD_DIM, kv0 + j * 16 + nc, 0,  half);
            s[j] = wmma16(qa0, kb0, s[j]);
            v16h kb1 = load_frag16(Kp, HEAD_DIM, kv0 + j * 16 + nc, 32, half);
            s[j] = wmma16(qa1, kb1, s[j]);
        }

        // ---- scale + causal mask + per-lane row max ----
        float mx[8];
        #pragma unroll
        for (int r = 0; r < 8; ++r) {
            const int qi = q0 + r + 8 * half;
            float mm = NEG_BIG;
            #pragma unroll
            for (int j = 0; j < 4; ++j) {
                const int kj = kv0 + j * 16 + nc;
                float v = s[j][r] * ATT_SCALE;
                v = (kj <= qi) ? v : NEG_BIG;
                s[j][r] = v;
                mm = fmaxf(mm, v);
            }
            mx[r] = mm;
        }
        // reduce max across the 16 lanes that hold one row (xor masks < 16)
        #pragma unroll
        for (int r = 0; r < 8; ++r) {
            #pragma unroll
            for (int off = 1; off < 16; off <<= 1)
                mx[r] = fmaxf(mx[r], __shfl_xor(mx[r], off, 32));
        }

        // ---- online softmax update ----
        float sc[8], rs[8];
        #pragma unroll
        for (int r = 0; r < 8; ++r) {
            const float mnew = fmaxf(mrun[r], mx[r]);
            sc[r]   = __expf(mrun[r] - mnew);
            mrun[r] = mnew;
            float srow = 0.0f;
            #pragma unroll
            for (int j = 0; j < 4; ++j) {
                const float p = __expf(s[j][r] - mnew);
                s[j][r] = p;
                srow += p;
            }
            rs[r] = srow;
        }
        #pragma unroll
        for (int r = 0; r < 8; ++r) {
            #pragma unroll
            for (int off = 1; off < 16; off <<= 1)
                rs[r] += __shfl_xor(rs[r], off, 32);
            lrun[r] = lrun[r] * sc[r] + rs[r];
        }
        #pragma unroll
        for (int t = 0; t < 4; ++t)
            #pragma unroll
            for (int r = 0; r < 8; ++r)
                acc[t][r] *= sc[r];

        // ---- stage P to LDS (D-layout -> row-major 16x64) ----
        #pragma unroll
        for (int j = 0; j < 4; ++j)
            #pragma unroll
            for (int r = 0; r < 8; ++r)
                pw[(r + 8 * half) * 64 + j * 16 + nc] = (_Float16)s[j][r];
        asm volatile("s_wait_dscnt 0" ::: "memory");   // RAW: all lane stores visible

        // ---- O += P V  (K-dim 64 in two 32 chunks; V^T gives contiguous B) ----
        #pragma unroll
        for (int kc = 0; kc < 64; kc += 32) {
            v16h pa = load_frag16(pw, 64, nc, kc, half);
            #pragma unroll
            for (int t = 0; t < 4; ++t) {
                v16h vb = load_frag16(Vp, SEQ_T, t * 16 + nc, kv0 + kc, half);
                acc[t] = wmma16(pa, vb, acc[t]);
            }
        }
        asm volatile("s_wait_dscnt 0" ::: "memory");   // WAR before next P store
    }

    // ---- normalize and store f16 [BT, D] ----
    float inv[8];
    #pragma unroll
    for (int r = 0; r < 8; ++r) inv[r] = 1.0f / lrun[r];
    #pragma unroll
    for (int t = 0; t < 4; ++t) {
        #pragma unroll
        for (int r = 0; r < 8; ++r) {
            const int row = q0 + r + 8 * half;
            AO[((size_t)b * SEQ_T + row) * D_MODEL + h * HEAD_DIM + t * 16 + nc] =
                (_Float16)(acc[t][r] * inv[r]);
        }
    }
}

// ---------------------------------------------------------------------------
extern "C" void kernel_launch(void* const* d_in, const int* in_sizes, int n_in,
                              void* d_out, int out_size, void* d_ws, size_t ws_size,
                              hipStream_t stream) {
    (void)in_sizes; (void)n_in; (void)out_size; (void)ws_size;

    const float* x  = (const float*)d_in[0];
    const float* Wq = (const float*)d_in[1];
    const float* bq = (const float*)d_in[2];
    const float* Wk = (const float*)d_in[3];
    const float* bk = (const float*)d_in[4];
    const float* Wv = (const float*)d_in[5];
    const float* bv = (const float*)d_in[6];
    const float* Wo = (const float*)d_in[7];
    const float* bo = (const float*)d_in[8];
    float* out = (float*)d_out;

    const size_t NX = (size_t)BT * D_MODEL;        // 4M halves
    const size_t NW = (size_t)D_MODEL * D_MODEL;   // 1M halves
    _Float16* xh  = (_Float16*)d_ws;
    _Float16* WTq = xh  + NX;
    _Float16* WTk = WTq + NW;
    _Float16* WTv = WTk + NW;
    _Float16* WTo = WTv + NW;
    _Float16* Qh  = WTo + NW;
    _Float16* Kh  = Qh  + NX;
    _Float16* VT  = Kh  + NX;
    _Float16* AO  = VT  + NX;

    // prep: convert + transpose weights
    mha_convert_x<<<NX / 256, 256, 0, stream>>>(x, xh);
    mha_transpose_w<<<dim3(NW / 256, 4), 256, 0, stream>>>(Wq, Wk, Wv, Wo,
                                                           WTq, WTk, WTv, WTo);

    // projections
    dim3 gg(BT / 128, D_MODEL / 64);
    mha_gemm<<<gg, 256, 0, stream>>>(xh, WTq, bq, 0, Qh, nullptr);
    mha_gemm<<<gg, 256, 0, stream>>>(xh, WTk, bk, 0, Kh, nullptr);
    mha_gemm<<<gg, 256, 0, stream>>>(xh, WTv, bv, 1, VT, nullptr);

    // flash attention
    mha_attention<<<dim3(SEQ_T / 128, HEADS, BATCH), 256, 0, stream>>>(Qh, Kh, VT, AO);

    // output projection (f32 + bias)
    mha_gemm<<<gg, 256, 0, stream>>>(AO, WTo, bo, 2, nullptr, out);
}
